// HiModel_27152783245981
// MI455X (gfx1250) — matrix operand order repaired and checked by reading, not compiled, not run on hardware
//
#include <hip/hip_runtime.h>
#include <math.h>

// ---------------------------------------------------------------------------
// HiModel fused pipeline for MI455X (gfx1250, wave32):
//   * all GEMMs on the WMMA bf16 tensor path (f32 accumulate)
//   * tiles fed by the Tensor Data Mover (tensor_load_to_lds, TENSORcnt),
//     double-buffered LDS, falling back to manual staging if the builtin is
//     not available on the current toolchain.
// ---------------------------------------------------------------------------

typedef __attribute__((ext_vector_type(16))) __bf16        v16bf;
typedef __attribute__((ext_vector_type(8)))  float         v8f;
typedef __attribute__((ext_vector_type(4)))  unsigned int  v4u;
typedef __attribute__((ext_vector_type(8)))  int           v8i;
typedef __attribute__((ext_vector_type(4)))  int           v4i;

static constexpr int BATCH = 2048;
static constexpr int NEI   = 45;
static constexpr int DT    = 768;
static constexpr int DV    = 1024;
static constexpr int MID   = 512;
static constexpr int FFN   = 2048;

#define DEVINL __device__ __forceinline__

// TDM available? (device pass only; host pass always takes fallback)
#if defined(__AMDGCN__) && defined(__has_builtin)
#if __has_builtin(__builtin_amdgcn_tensor_load_to_lds) && \
    __has_builtin(__builtin_amdgcn_s_wait_tensorcnt)
#define GEMM_TDM 1
#endif
#endif
#ifndef GEMM_TDM
#define GEMM_TDM 0
#endif

DEVINL unsigned short f2bf(float f) {
  unsigned u = __builtin_bit_cast(unsigned, f);
  u += 0x7FFFu + ((u >> 16) & 1u);           // round-to-nearest-even
  return (unsigned short)(u >> 16);
}

DEVINL float sigmoidf(float x) { return 1.f / (1.f + expf(-x)); }

#if GEMM_TDM
// ---------------------------------------------------------------------------
// Issue one 2-D TDM tile load: bf16 tensor, tile_d0 contiguous elements per
// row, tile_d1 rows, row stride `stride0` elements.  LDS destination gets
// 2 DWORDs of padding every 16 DWORDs (row = 32 bf16 -> LDS stride 36 shorts,
// bank-conflict-free for the WMMA fragment reads).
// ---------------------------------------------------------------------------
DEVINL void tdm_load_2d(unsigned lds_off, const unsigned short* gbase,
                        long long elem_off, int tile_d0, int tile_d1,
                        long long stride0) {
  const unsigned long long ga =
      (unsigned long long)(uintptr_t)gbase + (unsigned long long)elem_off * 2ull;
  v4u g0;
  g0[0] = 1u;                                              // count=1 (valid)
  g0[1] = lds_off;                                         // LDS byte address
  g0[2] = (unsigned)(ga & 0xFFFFFFFFull);                  // global_addr[31:0]
  g0[3] = (unsigned)((ga >> 32) & 0x01FFFFFFull) | (2u << 30); // [56:32] | type=2
  v8i g1;
  // data_size=1 (2B) | pad_enable | pad_interval=3 (16 DW) | pad_amount=1 (2 DW)
  g1[0] = (int)((1u << 16) | (1u << 20) | (3u << 22) | (1u << 25));
  const unsigned td0 = (unsigned)tile_d0;   // tensor dims == tile dims: no OOB
  const unsigned td1 = (unsigned)tile_d1;
  g1[1] = (int)((td0 & 0xFFFFu) << 16);                    // tensor_dim0[15:0]
  g1[2] = (int)(((td0 >> 16) & 0xFFFFu) | ((td1 & 0xFFFFu) << 16));
  g1[3] = (int)(((td1 >> 16) & 0xFFFFu) | (((unsigned)tile_d0 & 0xFFFFu) << 16));
  g1[4] = (int)((unsigned)tile_d1 & 0xFFFFu);              // tile_dim1, tile_dim2=0
  g1[5] = (int)(unsigned)(stride0 & 0xFFFFFFFFll);         // dim0_stride[31:0]
  g1[6] = (int)(unsigned)((stride0 >> 32) & 0xFFFFll);     // [47:32]; dim1_stride=0
  g1[7] = 0;
  const v4i z = {0, 0, 0, 0};
#if __clang_major__ >= 23
  const v8i z8 = {0, 0, 0, 0, 0, 0, 0, 0};
  __builtin_amdgcn_tensor_load_to_lds(g0, g1, z, z, z8, 0);
#else
  __builtin_amdgcn_tensor_load_to_lds(g0, g1, z, z, 0);
#endif
}
#endif  // GEMM_TDM

// ---------------------------------------------------------------------------
// Tiled GEMM: C[M,N] = act(A[M,K] @ B^T + bias),  A bf16 [M,K], B bf16 [N,K].
//   act: 0 = none, 1 = ReLU.  Optional bf16 dual store (feeds later GEMMs).
// Requires M%128==0, N%64==0, K%32==0 (true for every call in this model).
// Block = 256 threads = 8 wave32; tile 128x64x32; each wave owns 32x32 via
// 2x2 v_wmma_f32_16x16x32_bf16 accumulators.  Tiles arrive via TDM into
// double-buffered padded LDS; wave 0 drives the DMA pipeline.
// ---------------------------------------------------------------------------
__global__ __launch_bounds__(256)
void wmma_gemm_kernel(const unsigned short* __restrict__ A,
                      const unsigned short* __restrict__ B,
                      const float* __restrict__ bias, float* __restrict__ C,
                      unsigned short* __restrict__ Cbf,
                      int M, int N, int K, int act) {
  __shared__ unsigned short As[2][128 * 36];   // stride 36: conflict-free frags
  __shared__ unsigned short Bs[2][64 * 36];

  const int tid  = threadIdx.x;
  const int m0   = blockIdx.y * 128;
  const int n0   = blockIdx.x * 64;
  const int lane = tid & 31;
  const int wid  = tid >> 5;
  const int wm   = (wid >> 1) * 32;
  const int wn   = (wid & 1) * 32;
  const int l15  = lane & 15;
  const int hi   = lane >> 4;

  const v8f vzero = {0.f, 0.f, 0.f, 0.f, 0.f, 0.f, 0.f, 0.f};
  v8f acc[2][2];
  acc[0][0] = vzero; acc[0][1] = vzero; acc[1][0] = vzero; acc[1][1] = vzero;

  union Frag { unsigned int u[8]; v16bf v; };
  int cur = 0;

#if GEMM_TDM
  const unsigned aBase = (unsigned)(uintptr_t)&As[0][0];
  const unsigned bBase = (unsigned)(uintptr_t)&Bs[0][0];
  const unsigned aBytes = 128 * 36 * 2, bBytes = 64 * 36 * 2;
  if (wid == 0) {   // prologue: DMA first tile pair
    tdm_load_2d(aBase, A, (long long)m0 * K, 32, 128, K);
    tdm_load_2d(bBase, B, (long long)n0 * K, 32, 64, K);
  }
#endif

  for (int k0 = 0; k0 < K; k0 += 32) {
#if GEMM_TDM
    if (wid == 0) {
      if (k0 + 32 < K) {   // DMA next tile pair into the other buffer
        tdm_load_2d(aBase + (cur ^ 1) * aBytes, A,
                    (long long)m0 * K + (k0 + 32), 32, 128, K);
        tdm_load_2d(bBase + (cur ^ 1) * bBytes, B,
                    (long long)n0 * K + (k0 + 32), 32, 64, K);
        __builtin_amdgcn_s_wait_tensorcnt(2);  // current pair complete
      } else {
        __builtin_amdgcn_s_wait_tensorcnt(0);
      }
    }
#else
    // Fallback: manual staging of bf16 tiles
#pragma unroll
    for (int i = tid; i < 128 * 32; i += 256) {
      int r = i >> 5, c = i & 31;
      As[cur][r * 36 + c] = A[(size_t)(m0 + r) * K + (k0 + c)];
    }
#pragma unroll
    for (int i = tid; i < 64 * 32; i += 256) {
      int n = i >> 5, c = i & 31;
      Bs[cur][n * 36 + c] = B[(size_t)(n0 + n) * K + (k0 + c)];
    }
#endif
    __syncthreads();

    // ---- fragments per documented 16-bit wave32 layouts ----
    Frag fa[2], fb[2];
#pragma unroll
    for (int t = 0; t < 2; ++t) {
      const unsigned short* Ar = &As[cur][(wm + t * 16 + l15) * 36];
      const int kb = hi * 8;                 // lanes 16-31 start at K=8
#pragma unroll
      for (int j = 0; j < 8; ++j) {
        int k = kb + ((j < 4) ? (2 * j) : (16 + 2 * (j - 4)));
        fa[t].u[j] = *(const unsigned int*)(Ar + k);
      }
    }
#pragma unroll
    for (int s = 0; s < 2; ++s) {
      const unsigned short* Br = &Bs[cur][(wn + s * 16 + l15) * 36];
      const int kb = hi * 8;                 // lanes 16-31 start at K=8
#pragma unroll
      for (int j = 0; j < 8; ++j)
        fb[s].u[j] = *(const unsigned int*)(Br + kb + ((j < 4) ? (2 * j) : (16 + 2 * (j - 4))));
    }

#pragma unroll
    for (int t = 0; t < 2; ++t)
#pragma unroll
      for (int s = 0; s < 2; ++s)
        acc[t][s] = __builtin_amdgcn_wmma_f32_16x16x32_bf16(
            false, fa[t].v, false, fb[s].v, (short)0, acc[t][s], false, false);
    __syncthreads();   // all reads done before next DMA/stage overwrites
    cur ^= 1;
  }

  // ---- epilogue: C/D layout VGPR r -> M = r + 8*hi, N = l15 ----
#pragma unroll
  for (int t = 0; t < 2; ++t) {
#pragma unroll
    for (int s = 0; s < 2; ++s) {
      const int col = n0 + wn + s * 16 + l15;
      const float bv = bias ? bias[col] : 0.f;
#pragma unroll
      for (int r = 0; r < 8; ++r) {
        const int row = m0 + wm + t * 16 + hi * 8 + r;
        float v = acc[t][s][r] + bv;
        if (act == 1) v = v > 0.f ? v : 0.f;
        C[(size_t)row * N + col] = v;
        if (Cbf) Cbf[(size_t)row * N + col] = f2bf(v);
      }
    }
  }
}

// ---------------------------------------------------------------------------
// f32 -> bf16 copy (row-major), grid-stride.
// ---------------------------------------------------------------------------
__global__ __launch_bounds__(256)
void convert_kernel(const float* __restrict__ x, unsigned short* __restrict__ y,
                    int n) {
  for (int i = blockIdx.x * 256 + threadIdx.x; i < n; i += gridDim.x * 256)
    y[i] = f2bf(x[i]);
}

// ---------------------------------------------------------------------------
// [K,N] f32 -> [N,K] bf16 transpose (weights to A@B^T form). K,N % 32 == 0.
// ---------------------------------------------------------------------------
__global__ __launch_bounds__(256)
void transpose_convert_kernel(const float* __restrict__ B,
                              unsigned short* __restrict__ BT, int K, int N) {
  __shared__ float t[32][33];
  const int kb = blockIdx.y * 32, nb = blockIdx.x * 32;
  const int tx = threadIdx.x & 31, ty = threadIdx.x >> 5;   // 32 x 8
  for (int i = ty; i < 32; i += 8)
    t[i][tx] = B[(size_t)(kb + i) * N + (nb + tx)];
  __syncthreads();
  for (int i = ty; i < 32; i += 8)
    BT[(size_t)(nb + i) * K + (kb + tx)] = f2bf(t[tx][i]);
}

// ---------------------------------------------------------------------------
// Masked top-45 per row (matches jax.lax.top_k: descending, low-index ties).
// ---------------------------------------------------------------------------
__global__ __launch_bounds__(256)
void topk_kernel(const float* __restrict__ noise, const int* __restrict__ clu,
                 int* __restrict__ idx_out, float* __restrict__ valid_out) {
  const int b = blockIdx.x;
  __shared__ float s[BATCH];
  __shared__ float rv[256];
  __shared__ int   ri[256];
  const int tid = threadIdx.x;
  const int myclu = clu[b];

  for (int j = tid; j < BATCH; j += 256)
    s[j] = (clu[j] == myclu && j != b) ? noise[(size_t)b * BATCH + j] : -1e9f;
  __syncthreads();

  for (int it = 0; it < NEI; ++it) {
    float bv = -2e9f; int bi = 0;
    for (int j = tid; j < BATCH; j += 256) {
      float v = s[j];
      if (v > bv || (v == bv && j < bi)) { bv = v; bi = j; }
    }
    rv[tid] = bv; ri[tid] = bi;
    __syncthreads();
    for (int off = 128; off > 0; off >>= 1) {
      if (tid < off) {
        float ov = rv[tid + off]; int oi = ri[tid + off];
        if (ov > rv[tid] || (ov == rv[tid] && oi < ri[tid])) { rv[tid] = ov; ri[tid] = oi; }
      }
      __syncthreads();
    }
    if (tid == 0) {
      int sel = ri[0];
      idx_out[b * NEI + it]   = sel;
      valid_out[b * NEI + it] = (rv[0] > -1e8f) ? 1.f : 0.f;
      s[sel] = -3e9f;
    }
    __syncthreads();
  }
}

// ---------------------------------------------------------------------------
// Perceptron attention (gather rewrite):
//   e[n] = vp . tanh(qp[b] + bp + valid[n]*proj[idx[n]])
//   out  = sum_n softmax(e)[n]*valid[n]*tri[idx[n]]   (+ bf16 copy for GRU GEMM)
// ---------------------------------------------------------------------------
__global__ __launch_bounds__(256)
void attn_kernel(const float* __restrict__ qp, const float* __restrict__ proj,
                 const float* __restrict__ tri, const float* __restrict__ bp,
                 const float* __restrict__ vp, const int* __restrict__ idx,
                 const float* __restrict__ valid, float* __restrict__ out,
                 unsigned short* __restrict__ outbf, int D) {
  const int b = blockIdx.x;
  __shared__ float qb[1024];
  __shared__ float e[NEI];
  __shared__ float coef[NEI];
  __shared__ int   sidx[NEI];
  __shared__ float sw[NEI];
  const int tid = threadIdx.x;

  if (tid < NEI) { sidx[tid] = idx[b * NEI + tid]; sw[tid] = valid[b * NEI + tid]; }
  for (int d = tid; d < D; d += 256) qb[d] = qp[(size_t)b * D + d] + bp[d];
  __syncthreads();

  const int wid = tid >> 5, lane = tid & 31;
  for (int n = wid; n < NEI; n += 8) {
    const float w = sw[n];
    const float* pr = proj + (size_t)sidx[n] * D;
    float a = 0.f;
    for (int d = lane; d < D; d += 32)
      a += vp[d] * tanhf(qb[d] + w * pr[d]);
    for (int off = 16; off > 0; off >>= 1) a += __shfl_xor(a, off, 32);
    if (lane == 0) e[n] = a;
  }
  __syncthreads();

  if (tid == 0) {
    float m = -3.0e38f;
    for (int n = 0; n < NEI; ++n) m = fmaxf(m, e[n]);
    float sum = 0.f;
    for (int n = 0; n < NEI; ++n) { float x = expf(e[n] - m); coef[n] = x; sum += x; }
    const float inv = 1.f / sum;
    for (int n = 0; n < NEI; ++n) coef[n] *= inv * sw[n];
  }
  __syncthreads();

  for (int d = tid; d < D; d += 256) {
    float a = 0.f;
#pragma unroll 5
    for (int n = 0; n < NEI; ++n)
      a += coef[n] * tri[(size_t)sidx[n] * D + d];
    out[(size_t)b * D + d] = a;
    outbf[(size_t)b * D + d] = f2bf(a);
  }
}

// ---------------------------------------------------------------------------
// Train-mode BatchNorm over batch dim + ReLU (+ bf16 copy). Block per column.
// ---------------------------------------------------------------------------
__global__ __launch_bounds__(256)
void bn_relu_kernel(const float* __restrict__ x, const float* __restrict__ g,
                    const float* __restrict__ be, float* __restrict__ y,
                    unsigned short* __restrict__ ybf, int Bn, int C) {
  const int c = blockIdx.x;
  __shared__ float s1[256], s2[256];
  const int tid = threadIdx.x;
  float sum = 0.f, sq = 0.f;
  for (int r = tid; r < Bn; r += 256) {
    float v = x[(size_t)r * C + c];
    sum += v; sq += v * v;
  }
  s1[tid] = sum; s2[tid] = sq;
  __syncthreads();
  for (int off = 128; off > 0; off >>= 1) {
    if (tid < off) { s1[tid] += s1[tid + off]; s2[tid] += s2[tid + off]; }
    __syncthreads();
  }
  const float m   = s1[0] / (float)Bn;
  const float var = s2[0] / (float)Bn - m * m;
  const float inv = 1.f / sqrtf(var + 1e-5f);
  const float gc = g[c], bc = be[c];
  for (int r = tid; r < Bn; r += 256) {
    float v = gc * (x[(size_t)r * C + c] - m) * inv + bc;
    v = v > 0.f ? v : 0.f;
    y[(size_t)r * C + c] = v;
    ybf[(size_t)r * C + c] = f2bf(v);
  }
}

// ---------------------------------------------------------------------------
// GRU cell elementwise combine (+ optional bf16 copy).
// ---------------------------------------------------------------------------
__global__ __launch_bounds__(256)
void gru_kernel(const float* __restrict__ gi, const float* __restrict__ gh,
                const float* __restrict__ h, float* __restrict__ out,
                unsigned short* __restrict__ outbf) {
  const int i = blockIdx.x * 256 + threadIdx.x;
  if (i >= BATCH * MID) return;
  const int b = i >> 9;
  const int c = i & (MID - 1);
  const float* gib = gi + (size_t)b * 3 * MID;
  const float* ghb = gh + (size_t)b * 3 * MID;
  const float r = sigmoidf(gib[c]           + ghb[c]);
  const float z = sigmoidf(gib[MID + c]     + ghb[MID + c]);
  const float n = tanhf   (gib[2 * MID + c] + r * ghb[2 * MID + c]);
  const float v = (1.f - z) * n + z * h[i];
  out[i] = v;
  if (outbf) outbf[i] = f2bf(v);
}

// ---------------------------------------------------------------------------
// out = LayerNorm(a + b) over last dim (MID) (+ optional bf16 copy).
// ---------------------------------------------------------------------------
__global__ __launch_bounds__(256)
void add_ln_kernel(const float* __restrict__ a, const float* __restrict__ bsrc,
                   const float* __restrict__ g, const float* __restrict__ be,
                   float* __restrict__ out, unsigned short* __restrict__ outbf) {
  const int row = blockIdx.x;
  __shared__ float s[MID];
  __shared__ float red[256];
  const int tid = threadIdx.x;
  float lsum = 0.f;
  for (int c = tid; c < MID; c += 256) {
    float v = a[(size_t)row * MID + c] + bsrc[(size_t)row * MID + c];
    s[c] = v; lsum += v;
  }
  red[tid] = lsum; __syncthreads();
  for (int off = 128; off > 0; off >>= 1) {
    if (tid < off) red[tid] += red[tid + off];
    __syncthreads();
  }
  const float m = red[0] / (float)MID;
  __syncthreads();
  float lsq = 0.f;
  for (int c = tid; c < MID; c += 256) { float d = s[c] - m; lsq += d * d; }
  red[tid] = lsq; __syncthreads();
  for (int off = 128; off > 0; off >>= 1) {
    if (tid < off) red[tid] += red[tid + off];
    __syncthreads();
  }
  const float inv = 1.f / sqrtf(red[0] / (float)MID + 1e-5f);
  for (int c = tid; c < MID; c += 256) {
    const float v = g[c] * (s[c] - m) * inv + be[c];
    out[(size_t)row * MID + c] = v;
    if (outbf) outbf[(size_t)row * MID + c] = f2bf(v);
  }
}

// ---------------------------------------------------------------------------
// Classifier: softmax(x @ Wc + bc) with CLS=2.
// ---------------------------------------------------------------------------
__global__ __launch_bounds__(256)
void cls_kernel(const float* __restrict__ x, const float* __restrict__ Wc,
                const float* __restrict__ bc, float* __restrict__ out) {
  const int b = blockIdx.x * 256 + threadIdx.x;
  if (b >= BATCH) return;
  float a0 = bc[0], a1 = bc[1];
  const float* xb = x + (size_t)b * MID;
  for (int k = 0; k < MID; ++k) {
    const float v = xb[k];
    a0 += v * Wc[k * 2 + 0];
    a1 += v * Wc[k * 2 + 1];
  }
  const float m = fmaxf(a0, a1);
  const float e0 = expf(a0 - m), e1 = expf(a1 - m);
  const float inv = 1.f / (e0 + e1);
  out[(size_t)b * 2 + 0] = e0 * inv;
  out[(size_t)b * 2 + 1] = e1 * inv;
}

// ---------------------------------------------------------------------------
// Host orchestration
// ---------------------------------------------------------------------------
extern "C" void kernel_launch(void* const* d_in, const int* in_sizes, int n_in,
                              void* d_out, int out_size, void* d_ws, size_t ws_size,
                              hipStream_t stream) {
  (void)in_sizes; (void)n_in; (void)out_size; (void)ws_size;

  const float* input_t  = (const float*)d_in[0];
  const float* input_v  = (const float*)d_in[1];
  const int*   te_clu   = (const int*)  d_in[2];
  const int*   im_clu   = (const int*)  d_in[3];
  const float* t_tri    = (const float*)d_in[4];
  const float* v_tri    = (const float*)d_in[5];
  const float* noise_t  = (const float*)d_in[6];
  const float* noise_v  = (const float*)d_in[7];
  const float* Wq_t = (const float*)d_in[8],  *Wk_t = (const float*)d_in[9];
  const float* bp_t = (const float*)d_in[10], *vp_t = (const float*)d_in[11];
  const float* Wq_v = (const float*)d_in[12], *Wk_v = (const float*)d_in[13];
  const float* bp_v = (const float*)d_in[14], *vp_v = (const float*)d_in[15];
  const float* W_lt = (const float*)d_in[16], *b_lt = (const float*)d_in[17];
  const float* g_bt = (const float*)d_in[18], *be_bt = (const float*)d_in[19];
  const float* W_lv = (const float*)d_in[20], *b_lv = (const float*)d_in[21];
  const float* g_bv = (const float*)d_in[22], *be_bv = (const float*)d_in[23];
  const float* Wih_t = (const float*)d_in[24], *Whh_t = (const float*)d_in[25];
  const float* bih_t = (const float*)d_in[26], *bhh_t = (const float*)d_in[27];
  const float* Wih_v = (const float*)d_in[28], *Whh_v = (const float*)d_in[29];
  const float* bih_v = (const float*)d_in[30], *bhh_v = (const float*)d_in[31];
  const float* Wv_f  = (const float*)d_in[32], *Wo_f  = (const float*)d_in[33];
  const float* g_ln1 = (const float*)d_in[34], *b_ln1 = (const float*)d_in[35];
  const float* Wf1 = (const float*)d_in[36], *bf1 = (const float*)d_in[37];
  const float* Wf2 = (const float*)d_in[38], *bf2 = (const float*)d_in[39];
  const float* g_ln2 = (const float*)d_in[40], *b_ln2 = (const float*)d_in[41];
  const float* Wc = (const float*)d_in[42], *bc = (const float*)d_in[43];

  // ---- output slices (tuple concat, all f32) ----
  float* out = (float*)d_out;
  const size_t o_ttri = 0;
  const size_t o_vtri = o_ttri + (size_t)BATCH * DT;
  const size_t o_ht   = o_vtri + (size_t)BATCH * DV;
  const size_t o_hv   = o_ht   + (size_t)BATCH * MID;
  const size_t o_grt  = o_hv   + (size_t)BATCH * MID;
  const size_t o_grv  = o_grt  + (size_t)BATCH * MID;
  const size_t o_mm   = o_grv  + (size_t)BATCH * MID;
  const size_t o_cls  = o_mm   + (size_t)BATCH * MID;
  float* ht  = out + o_ht;
  float* hv  = out + o_hv;
  float* grt = out + o_grt;
  float* grv = out + o_grv;
  float* mm  = out + o_mm;

  // ---- workspace bump allocator (bytes) ----
  char* wsb = (char*)d_ws;
  size_t cur = 0;
  auto allocf = [&](size_t n) -> float* {
    float* p = (float*)(wsb + cur); cur += ((n * 4 + 255) & ~(size_t)255); return p;
  };
  auto alloch = [&](size_t n) -> unsigned short* {
    unsigned short* p = (unsigned short*)(wsb + cur);
    cur += ((n * 2 + 255) & ~(size_t)255); return p;
  };

  int*   idxT = (int*)allocf((size_t)BATCH * NEI);
  float* valT = allocf((size_t)BATCH * NEI);
  int*   idxV = (int*)allocf((size_t)BATCH * NEI);
  float* valV = allocf((size_t)BATCH * NEI);
  float* qpT   = allocf((size_t)BATCH * DT);
  float* projT = allocf((size_t)BATCH * DT);
  float* qpV   = allocf((size_t)BATCH * DV);
  float* projV = allocf((size_t)BATCH * DV);
  float* embT  = allocf((size_t)BATCH * DT);
  float* embV  = allocf((size_t)BATCH * DV);
  float* linT  = allocf((size_t)BATCH * MID);
  float* linV  = allocf((size_t)BATCH * MID);
  float* giT   = allocf((size_t)BATCH * 3 * MID);
  float* ghT   = allocf((size_t)BATCH * 3 * MID);
  float* giV   = allocf((size_t)BATCH * 3 * MID);
  float* ghV   = allocf((size_t)BATCH * 3 * MID);
  float* tmp1  = allocf((size_t)BATCH * MID);
  float* tmp2  = allocf((size_t)BATCH * MID);
  float* h1    = allocf((size_t)BATCH * MID);
  float* ffn1  = allocf((size_t)BATCH * FFN);
  float* ffn2  = allocf((size_t)BATCH * MID);

  // bf16 activations (A operands)
  unsigned short* tbfT   = alloch((size_t)BATCH * DT);
  unsigned short* tbfV   = alloch((size_t)BATCH * DV);
  unsigned short* inbfT  = alloch((size_t)BATCH * DT);
  unsigned short* inbfV  = alloch((size_t)BATCH * 4096);
  unsigned short* embbfT = alloch((size_t)BATCH * DT);
  unsigned short* embbfV = alloch((size_t)BATCH * DV);
  unsigned short* htbf   = alloch((size_t)BATCH * MID);
  unsigned short* hvbf   = alloch((size_t)BATCH * MID);
  unsigned short* grtbf  = alloch((size_t)BATCH * MID);
  unsigned short* tmp1bf = alloch((size_t)BATCH * MID);
  unsigned short* h1bf   = alloch((size_t)BATCH * MID);
  unsigned short* ffn1bf = alloch((size_t)BATCH * FFN);

  // bf16 weights, all as [N,K]
  unsigned short* bWq_t  = alloch((size_t)DT * DT);
  unsigned short* bWk_t  = alloch((size_t)DT * DT);
  unsigned short* bWq_v  = alloch((size_t)DV * DV);
  unsigned short* bWk_v  = alloch((size_t)DV * DV);
  unsigned short* bW_lt  = alloch((size_t)MID * DT);
  unsigned short* bW_lv  = alloch((size_t)MID * 4096);
  unsigned short* bWih_t = alloch((size_t)3 * MID * DT);
  unsigned short* bWhh_t = alloch((size_t)3 * MID * MID);
  unsigned short* bWih_v = alloch((size_t)3 * MID * DV);
  unsigned short* bWhh_v = alloch((size_t)3 * MID * MID);
  unsigned short* bWv_f  = alloch((size_t)MID * MID);
  unsigned short* bWo_f  = alloch((size_t)MID * MID);
  unsigned short* bWf1   = alloch((size_t)FFN * MID);
  unsigned short* bWf2   = alloch((size_t)MID * FFN);

  auto conv = [&](const float* x, unsigned short* y, size_t n) {
    convert_kernel<<<4096, 256, 0, stream>>>(x, y, (int)n);
  };
  auto tconv = [&](const float* Bm, unsigned short* BT, int K, int N) {
    transpose_convert_kernel<<<dim3(N / 32, K / 32), 256, 0, stream>>>(Bm, BT, K, N);
  };
  auto gemm = [&](const unsigned short* Abf, const unsigned short* Bbf,
                  const float* bias, float* C, unsigned short* Cbf,
                  int M, int N, int K, int act) {
    dim3 grid((unsigned)(N / 64), (unsigned)(M / 128));
    wmma_gemm_kernel<<<grid, 256, 0, stream>>>(Abf, Bbf, bias, C, Cbf, M, N, K, act);
  };

  // ---- passthrough outputs ----
  hipMemcpyAsync(out + o_ttri, t_tri, (size_t)BATCH * DT * sizeof(float),
                 hipMemcpyDeviceToDevice, stream);
  hipMemcpyAsync(out + o_vtri, v_tri, (size_t)BATCH * DV * sizeof(float),
                 hipMemcpyDeviceToDevice, stream);

  // ---- precision prep: bf16 activations + transposed bf16 weights ----
  conv(t_tri, tbfT, (size_t)BATCH * DT);
  conv(v_tri, tbfV, (size_t)BATCH * DV);
  conv(input_t, inbfT, (size_t)BATCH * DT);
  conv(input_v, inbfV, (size_t)BATCH * 4096);
  tconv(Wq_t, bWq_t, DT, DT);   tconv(Wk_t, bWk_t, DT, DT);
  tconv(Wq_v, bWq_v, DV, DV);   tconv(Wk_v, bWk_v, DV, DV);
  tconv(W_lt, bW_lt, DT, MID);  tconv(W_lv, bW_lv, 4096, MID);
  conv(Wih_t, bWih_t, (size_t)3 * MID * DT);    // already [N,K]
  conv(Whh_t, bWhh_t, (size_t)3 * MID * MID);
  conv(Wih_v, bWih_v, (size_t)3 * MID * DV);
  conv(Whh_v, bWhh_v, (size_t)3 * MID * MID);
  tconv(Wv_f, bWv_f, MID, MID); tconv(Wo_f, bWo_f, MID, MID);
  tconv(Wf1, bWf1, MID, FFN);   tconv(Wf2, bWf2, FFN, MID);

  // ---- neighbor selection ----
  topk_kernel<<<BATCH, 256, 0, stream>>>(noise_t, te_clu, idxT, valT);
  topk_kernel<<<BATCH, 256, 0, stream>>>(noise_v, im_clu, idxV, valV);

  // ---- attention projections (project once, gather rows) ----
  gemm(tbfT, bWq_t, nullptr, qpT,   nullptr, BATCH, DT, DT, 0);
  gemm(tbfT, bWk_t, nullptr, projT, nullptr, BATCH, DT, DT, 0);
  gemm(tbfV, bWq_v, nullptr, qpV,   nullptr, BATCH, DV, DV, 0);
  gemm(tbfV, bWk_v, nullptr, projV, nullptr, BATCH, DV, DV, 0);
  attn_kernel<<<BATCH, 256, 0, stream>>>(qpT, projT, t_tri, bp_t, vp_t,
                                         idxT, valT, embT, embbfT, DT);
  attn_kernel<<<BATCH, 256, 0, stream>>>(qpV, projV, v_tri, bp_v, vp_v,
                                         idxV, valV, embV, embbfV, DV);

  // ---- dense branch: Linear -> BN(train) -> ReLU ----
  gemm(inbfT, bW_lt, b_lt, linT, nullptr, BATCH, MID, DT, 0);
  gemm(inbfV, bW_lv, b_lv, linV, nullptr, BATCH, MID, 4096, 0);
  bn_relu_kernel<<<MID, 256, 0, stream>>>(linT, g_bt, be_bt, ht, htbf, BATCH, MID);
  bn_relu_kernel<<<MID, 256, 0, stream>>>(linV, g_bv, be_bv, hv, hvbf, BATCH, MID);

  // ---- GRU cells ----
  gemm(embbfT, bWih_t, bih_t, giT, nullptr, BATCH, 3 * MID, DT, 0);
  gemm(htbf,   bWhh_t, bhh_t, ghT, nullptr, BATCH, 3 * MID, MID, 0);
  gemm(embbfV, bWih_v, bih_v, giV, nullptr, BATCH, 3 * MID, DV, 0);
  gemm(hvbf,   bWhh_v, bhh_v, ghV, nullptr, BATCH, 3 * MID, MID, 0);
  gru_kernel<<<(BATCH * MID) / 256, 256, 0, stream>>>(giT, ghT, ht, grt, grtbf);
  gru_kernel<<<(BATCH * MID) / 256, 256, 0, stream>>>(giV, ghV, hv, grv, nullptr);

  // ---- cross-attention fusion + FFN ----
  gemm(grtbf,  bWv_f, nullptr, tmp1, tmp1bf, BATCH, MID, MID, 0);
  gemm(tmp1bf, bWo_f, nullptr, tmp2, nullptr, BATCH, MID, MID, 0);
  add_ln_kernel<<<BATCH, 256, 0, stream>>>(grv, tmp2, g_ln1, b_ln1, h1, h1bf);
  gemm(h1bf,   bWf1, bf1, ffn1, ffn1bf, BATCH, FFN, MID, 1);   // ReLU fused
  gemm(ffn1bf, bWf2, bf2, ffn2, nullptr, BATCH, MID, FFN, 0);
  add_ln_kernel<<<BATCH, 256, 0, stream>>>(h1, ffn2, g_ln2, b_ln2, mm, nullptr);

  // ---- classifier softmax ----
  cls_kernel<<<(BATCH + 255) / 256, 256, 0, stream>>>(mm, Wc, bc, out + o_cls);
}
